// MixerHead_83038897701601
// MI455X (gfx1250) — matrix operand order repaired and checked by reading, not compiled, not run on hardware
//
#include <hip/hip_runtime.h>

// ---------------------------------------------------------------------------
// MixerHead on MI455X (gfx1250).
//   1) softmax_rows : W = softmax(tril(conv_w))  -> bf16 [f][d]   (2 MB)
//   2) build_weff   : w_eff[o][m] = sum_h out_w  -> bf16 [o][m]   (2 MB)
//   3) transpose_cv : Xt[b][m][d] = bf16(x[b,d,m])  via TDM tensor_load_to_lds
//   4) gemm_nt<bf16,rowbias> : Y = W @ X + conv_b        (bf16 out)
//   5) gemm_nt<f32, colbias> : Out = Y @ w_eff^T + out_b (f32 out)
// GEMMs: pure global b128 fragment loads (A row-major [M][K], B N-major
// [N][K]), wave tile 32x64 (8 accumulators, 8 WMMA : 12 loads per K-step),
// block tile 128x128, f32 accumulate.
// ---------------------------------------------------------------------------

typedef __bf16  v8bf  __attribute__((ext_vector_type(8)));
typedef __bf16  v16bf __attribute__((ext_vector_type(16)));
typedef float   v8f   __attribute__((ext_vector_type(8)));
typedef unsigned int u32x4 __attribute__((ext_vector_type(4)));
typedef int          i32x4 __attribute__((ext_vector_type(4)));
typedef int          i32x8 __attribute__((ext_vector_type(8)));

constexpr int kB  = 16;
constexpr int kL  = 1024;
constexpr int kD  = 1024;
constexpr int kNH = 8;

static __device__ __forceinline__ unsigned short bf16_bits(float f) {
  union { float f; unsigned u; } a; a.f = f;
  unsigned u = a.u + 0x7FFFu + ((a.u >> 16) & 1u);   // RNE
  return (unsigned short)(u >> 16);
}
static __device__ __forceinline__ v16bf cat8(v8bf lo, v8bf hi) {
  return __builtin_shufflevector(lo, hi, 0,1,2,3,4,5,6,7,8,9,10,11,12,13,14,15);
}
static __device__ __forceinline__ v8f wmma_bf16(v16bf a, v16bf b, v8f c) {
  return __builtin_amdgcn_wmma_f32_16x16x32_bf16(false, a, false, b,
                                                 (short)0, c, false, false);
}
static __device__ __forceinline__ unsigned rfl(unsigned v) {
  return (unsigned)__builtin_amdgcn_readfirstlane((int)v);
}

// ---------------------------------------------------------------------------
// Kernel 1: W = softmax(tril(conv_w)) row-wise, bf16. The un-masked tril means
// every d>row slot contributes exp(0)=1 to Z and gets value 1/Z (faithful).
// ---------------------------------------------------------------------------
__global__ __launch_bounds__(256)
void softmax_rows(const float* __restrict__ conv_w, unsigned short* __restrict__ Wbf) {
  __shared__ float red[256];
  const int row = blockIdx.x;
  const int tid = threadIdx.x;
  const float* wr = conv_w + (size_t)row * kL;

  float part = 0.f;
  for (int i = tid; i < kL; i += 256)
    part += (i <= row) ? __expf(wr[i]) : 1.0f;
  red[tid] = part;
  __syncthreads();
  for (int s = 128; s > 0; s >>= 1) {
    if (tid < s) red[tid] += red[tid + s];
    __syncthreads();
  }
  const float inv = 1.0f / red[0];

  unsigned short* orow = Wbf + (size_t)row * kL;
  for (int i = tid; i < kL; i += 256)
    orow[i] = bf16_bits(((i <= row) ? __expf(wr[i]) : 1.0f) * inv);
}

// ---------------------------------------------------------------------------
// Kernel 2: w_eff[o][m] = sum_h out_w[o][h*D+m], bf16 row-major [o][m]
// (exactly the N-major layout GEMM2's B fragments want).
// ---------------------------------------------------------------------------
__global__ __launch_bounds__(256)
void build_weff(const float* __restrict__ out_w, unsigned short* __restrict__ weff) {
  const int idx = blockIdx.x * 256 + threadIdx.x;       // o*1024 + m
  const int o = idx >> 10, m = idx & 1023;
  const float* p = out_w + (size_t)o * (kNH * kD) + m;
  float s = 0.f;
#pragma unroll
  for (int h = 0; h < kNH; ++h) s += p[h * kD];
  weff[idx] = bf16_bits(s);
}

// ---------------------------------------------------------------------------
// Kernel 3: Xt[b][m][d] = bf16(x[b][d][m]).  64x64 f32 tile per workgroup,
// DMA'd into LDS by the Tensor Data Mover with pad_enable (1 DWORD pad every
// 64 DWORDs -> 65-DWORD row pitch, bank-conflict-free column reads), then
// transposed/converted out with coalesced bf16 stores.
// ---------------------------------------------------------------------------
__global__ __launch_bounds__(256)
void transpose_convert(const float* __restrict__ x, unsigned short* __restrict__ Xt) {
  __shared__ float tile[64 * 65];
  const int b  = blockIdx.z;
  const int k0 = blockIdx.y * 64;   // token (d) rows
  const int n0 = blockIdx.x * 64;   // model (m) cols
  const int tid = threadIdx.x;
  const float* src = x + ((size_t)b << 20) + (size_t)k0 * kD + n0;

#if __has_builtin(__builtin_amdgcn_tensor_load_to_lds)
  if (tid < 32) {                                    // one wave issues the DMA
    const unsigned long long ga = (unsigned long long)(const void*)src;
    const unsigned lds = (unsigned)(size_t)(void*)tile;  // low32 = LDS offset
    u32x4 g0;
    g0[0] = rfl(1u);                                 // count=1, user mode
    g0[1] = rfl(lds);                                // lds_addr
    g0[2] = rfl((unsigned)ga);                       // global_addr[31:0]
    g0[3] = rfl(((unsigned)(ga >> 32) & 0x01FFFFFFu) | (2u << 30)); // [56:32]|type=2
    i32x8 g1;
    // w0: data_size=4B(2), pad_enable=1, pad_interval=64dw(5), pad_amount=1dw(0)
    g1[0] = (int)rfl((2u << 16) | (1u << 20) | (5u << 22));
    g1[1] = (int)rfl((unsigned)(kD & 0xFFFF) << 16);            // tensor_dim0 lo16
    g1[2] = (int)rfl(((unsigned)kD >> 16) | ((unsigned)(kL & 0xFFFF) << 16)); // dim0 hi | dim1 lo
    g1[3] = (int)rfl(((unsigned)kL >> 16) | (64u << 16));       // dim1 hi | tile_dim0=64
    g1[4] = (int)rfl(64u);                                      // tile_dim1=64, tile_dim2=0
    g1[5] = (int)rfl((unsigned)kD);                             // dim0_stride lo32
    g1[6] = (int)rfl(0u);
    g1[7] = (int)rfl(0u);
    i32x4 z4 = {};
#if defined(__clang_major__) && (__clang_major__ >= 23)
    i32x8 z8 = {};
    __builtin_amdgcn_tensor_load_to_lds(g0, g1, z4, z4, z8, 0);
#else
    __builtin_amdgcn_tensor_load_to_lds(g0, g1, z4, z4, 0);
#endif
    __builtin_amdgcn_s_wait_tensorcnt(0);
  }
  __syncthreads();
#else
#pragma unroll
  for (int i = 0; i < 16; ++i) {
    const int idx = tid + 256 * i;
    const int kk = idx >> 6, nn = idx & 63;
    tile[kk * 65 + nn] = src[(size_t)kk * kD + nn];
  }
  __syncthreads();
#endif

  // transpose out: n fixed per 64 threads, lanes sweep consecutive k
  unsigned short* dst = Xt + ((size_t)b << 20);
  const int k = tid & 63;
#pragma unroll
  for (int i = 0; i < 16; ++i) {
    const int n = (tid >> 6) * 16 + i;
    dst[(size_t)(n0 + n) * kL + k0 + k] = bf16_bits(tile[k * 65 + n]);
  }
}

// ---------------------------------------------------------------------------
// Unified GEMM:  C[b] = A (MxK, row-major bf16) x Bt^T ([N][K] bf16) + bias,
// f32 accumulate.  Block = 8 waves (4M x 2N) = 128M x 128N; wave = 32M x 64N:
// 2 A-frags x 4 B-frags -> 8 WMMA per 12 b128 loads per K=32 step.
// Fragment layouts per CDNA5 ISA:
//   A lane(l%16, q=l/16): K = 8q+0..7 (elems 0-7), 16+8q+0..7 (elems 8-15)
//   B lane(n=l%16, h=l/16): K = 8h+0..7 (elems 0-7), 16+8h+0..7 (elems 8-15) at fixed column n
//   C VGPR r: row 8*(l/16)+r, col l%16
// ---------------------------------------------------------------------------
template<bool STORE_BF16, bool ROW_BIAS>
__global__ __launch_bounds__(256)
void gemm_nt(const unsigned short* __restrict__ A, const unsigned short* __restrict__ Bt,
             const float* __restrict__ bias, void* __restrict__ Cv,
             int aBatch, int bBatch) {
  const int b = blockIdx.z;
  const int tile_m0 = blockIdx.y * 128;
  const int tile_n0 = blockIdx.x * 128;
  const int tid  = threadIdx.x;
  const int lane = tid & 31, wave = tid >> 5;
  const int lrow = lane & 15, lhalf = lane >> 4;
  const int wm = wave >> 1, wn = wave & 1;
  const int row0 = tile_m0 + 32 * wm;
  const int col0 = tile_n0 + 64 * wn;

  const unsigned short* ar0 = A  + (size_t)b * (size_t)aBatch + (size_t)(row0 + lrow) * kD;
  const unsigned short* ar1 = ar0 + (size_t)16 * kD;
  const unsigned short* br[4];
#pragma unroll
  for (int j = 0; j < 4; ++j)
    br[j] = Bt + (size_t)b * (size_t)bBatch + (size_t)(col0 + 16 * j + lrow) * kD;

  v8f acc0[4] = {{}, {}, {}, {}};
  v8f acc1[4] = {{}, {}, {}, {}};
#pragma unroll 2
  for (int k0 = 0; k0 < kD; k0 += 32) {
    const int ao = k0 + 8 * lhalf;
    const int bo = k0 + 8 * lhalf;
    v16bf a0 = cat8(*(const v8bf*)(ar0 + ao), *(const v8bf*)(ar0 + ao + 16));
    v16bf a1 = cat8(*(const v8bf*)(ar1 + ao), *(const v8bf*)(ar1 + ao + 16));
    if (k0 + 32 < kD) {                               // global_prefetch_b8, WGP scope
      __builtin_prefetch(ar0 + k0 + 32, 0, 3);
      __builtin_prefetch(br[0] + k0 + 32, 0, 3);
    }
#pragma unroll
    for (int j = 0; j < 4; ++j) {
      v16bf bf = cat8(*(const v8bf*)(br[j] + bo), *(const v8bf*)(br[j] + bo + 16));
      acc0[j] = wmma_bf16(a0, bf, acc0[j]);
      acc1[j] = wmma_bf16(a1, bf, acc1[j]);
    }
  }

#pragma unroll
  for (int j = 0; j < 4; ++j) {
    const int c = col0 + 16 * j + lrow;
    const float cb = ROW_BIAS ? 0.f : bias[c];
#pragma unroll
    for (int r = 0; r < 8; ++r) {
      const int mA = row0 + 8 * lhalf + r;
      const int mB = mA + 16;
      const float v0 = acc0[j][r] + (ROW_BIAS ? bias[mA] : cb);
      const float v1 = acc1[j][r] + (ROW_BIAS ? bias[mB] : cb);
      if constexpr (STORE_BF16) {
        unsigned short* C = (unsigned short*)Cv + ((size_t)b << 20);
        C[(size_t)mA * kD + c] = bf16_bits(v0);
        C[(size_t)mB * kD + c] = bf16_bits(v1);
      } else {
        float* C = (float*)Cv + ((size_t)b << 20);
        C[(size_t)mA * kD + c] = v0;
        C[(size_t)mB * kD + c] = v1;
      }
    }
  }
}

// ---------------------------------------------------------------------------
extern "C" void kernel_launch(void* const* d_in, const int* in_sizes, int n_in,
                              void* d_out, int out_size, void* d_ws, size_t ws_size,
                              hipStream_t stream) {
  (void)in_sizes; (void)n_in; (void)out_size; (void)ws_size;
  const float* x      = (const float*)d_in[0];   // [16,1024,1024]
  const float* conv_w = (const float*)d_in[1];   // [1024,1024]
  const float* conv_b = (const float*)d_in[2];   // [1024]
  const float* out_w  = (const float*)d_in[3];   // [1024,8192]
  const float* out_b  = (const float*)d_in[4];   // [1024]
  float* out = (float*)d_out;                    // [16,1024,1024] f32

  char* ws = (char*)d_ws;
  unsigned short* Wbf  = (unsigned short*)(ws);                     //  2 MB
  unsigned short* weff = (unsigned short*)(ws + (1ull << 21));      //  2 MB
  unsigned short* Ybf  = (unsigned short*)(ws + (1ull << 22));      // 32 MB
  unsigned short* Xt   = (unsigned short*)(ws + (9ull << 22));      // 32 MB
  // total ws use: 68 MB

  softmax_rows<<<kL, 256, 0, stream>>>(conv_w, Wbf);
  build_weff<<<(kD * kD) / 256, 256, 0, stream>>>(out_w, weff);
  transpose_convert<<<dim3(kD / 64, kL / 64, kB), 256, 0, stream>>>(x, Xt);

  dim3 grid(kD / 128, kL / 128, kB);
  // Y[b] = W @ X[b] + conv_b(row): A = Wbf (shared), Bt = Xt[b]
  gemm_nt<true,  true ><<<grid, 256, 0, stream>>>(Wbf, Xt, conv_b, Ybf, 0, 1 << 20);
  // Out[b] = Y[b] @ w_eff^T + out_b(col): A = Ybf[b], Bt = weff (shared)
  gemm_nt<false, false><<<grid, 256, 0, stream>>>(Ybf, weff, out_b, out, 1 << 20, 0);
}
